// CapsuleNetwork_58506044506705
// MI455X (gfx1250) — matrix operand order repaired and checked by reading, not compiled, not run on hardware
//
#include <hip/hip_runtime.h>

// Problem constants (from reference)
#define Bn 1024
#define Ln 200
#define Dn 64
#define Kn 4
#define En 256   // K*D

typedef __attribute__((ext_vector_type(16))) __bf16 v16bf;
typedef __attribute__((ext_vector_type(8)))  float  v8f;

__device__ __forceinline__ float bf16u_to_f(unsigned short h) {
    unsigned int u = ((unsigned int)h) << 16;
    return __builtin_bit_cast(float, u);
}
__device__ __forceinline__ unsigned short f_to_bf16u(float f) {
    __bf16 h = (__bf16)f;   // RNE convert via v_cvt
    return __builtin_bit_cast(unsigned short, h);
}

// ---------------------------------------------------------------------------
// Kernel 1: per-position GEMM  hat[l, b, e] = sum_d seq[b,l,d] * W[l,e,d]
// bf16 WMMA (f32 accumulate). Grid: (B/16, L). Block: 256 threads = 8 waves.
// Each wave: one 16(b) x 16(e) tile pair, K=64 split into two x32 WMMA steps.
// hat stored bf16 in workspace, layout [L][B][En] (coalesced rows for kernel 2).
// ---------------------------------------------------------------------------
__global__ __launch_bounds__(256) void hat_gemm_kernel(
    const float* __restrict__ seq,      // [B, L, D]
    const float* __restrict__ W,        // [L, En, D]
    unsigned short* __restrict__ hat)   // [L, B, En] bf16
{
    __shared__ float A[16][65];         // padded: gather is bank-conflict free

    const int l   = blockIdx.y;
    const int b0  = blockIdx.x << 4;
    const int tid = threadIdx.x;
    const int wave = tid >> 5, lane = tid & 31;

    const int m   = lane & 15;          // A row (M)
    const int klo = (lane >> 4) << 3;   // A K sub-base per documented layout
    const int kbl = (lane >> 4) << 4;   // B K sub-base (lanes 16-31 -> K+16)
    const int n   = lane & 15;          // B/C column (N)

    const int e0 = ((wave << 1) + 0) * 16 + n;
    const int e1 = ((wave << 1) + 1) * 16 + n;
    const float* w0 = W + ((size_t)l * En + e0) * Dn;
    const float* w1 = W + ((size_t)l * En + e1) * Dn;

    // Warm L2/L0 for the weight rows while we wait on the LDS fill barrier.
    __builtin_prefetch(w0, 0, 0);       // -> global_prefetch_b8
    __builtin_prefetch(w1, 0, 0);

    // Cooperative load of the 16x64 f32 A tile (seq[b0..b0+15, l, :])
    {
        const int row = tid >> 4;
        const int c4  = (tid & 15) << 2;
        const float4 v = *(const float4*)(seq + ((size_t)(b0 + row) * Ln + l) * Dn + c4);
        A[row][c4 + 0] = v.x; A[row][c4 + 1] = v.y;
        A[row][c4 + 2] = v.z; A[row][c4 + 3] = v.w;
    }
    __syncthreads();

    v8f c0 = {}, c1 = {};
    #pragma unroll
    for (int ks = 0; ks < 2; ++ks) {
        const int kstep = ks << 5;      // 0, 32

        // A fragment: contiguous k = kstep + 16*(lane/16) + i, same k order as B
        v16bf a;
        #pragma unroll
        for (int i = 0; i < 16; ++i) {
            const int k = kstep + kbl + i;
            a[i] = (__bf16)A[m][k];
        }

        // B fragments: 16 consecutive f32 along d from each W row -> bf16
        const float4* q0 = (const float4*)(w0 + kstep + kbl);
        const float4* q1 = (const float4*)(w1 + kstep + kbl);
        v16bf bA, bB;
        #pragma unroll
        for (int j = 0; j < 4; ++j) {
            const float4 f0 = q0[j], f1 = q1[j];
            bA[j*4+0] = (__bf16)f0.x; bA[j*4+1] = (__bf16)f0.y;
            bA[j*4+2] = (__bf16)f0.z; bA[j*4+3] = (__bf16)f0.w;
            bB[j*4+0] = (__bf16)f1.x; bB[j*4+1] = (__bf16)f1.y;
            bB[j*4+2] = (__bf16)f1.z; bB[j*4+3] = (__bf16)f1.w;
        }

        c0 = __builtin_amdgcn_wmma_f32_16x16x32_bf16(false, a, false, bA,
                                                     (short)0, c0, false, false);
        c1 = __builtin_amdgcn_wmma_f32_16x16x32_bf16(false, a, false, bB,
                                                     (short)0, c1, false, false);
    }

    // Scatter C (16x16 f32 layout: lane<16 -> M=v, lane>=16 -> M=v+8) as bf16
    #pragma unroll
    for (int v = 0; v < 8; ++v) {
        const int brow = b0 + v + ((lane >> 4) << 3);
        const size_t base = ((size_t)l * Bn + brow) * En;
        hat[base + e0] = f_to_bf16u(c0[v]);
        hat[base + e1] = f_to_bf16u(c1[v]);
    }
}

// ---------------------------------------------------------------------------
// Kernel 2: dynamic routing, one workgroup per batch element.
// hat[b] (K*L*D bf16 = 100 KB) is DMA'd into LDS via the CDNA5 async
// global->LDS path (GLOBAL_LOAD_ASYNC_TO_LDS_B128, ASYNCcnt), then all
// 3 routing iterations run entirely out of LDS.
// ---------------------------------------------------------------------------
__global__ __launch_bounds__(256) void routing_kernel(
    const unsigned short* __restrict__ hat,   // [L, B, En] bf16
    const float* __restrict__ cwg,            // [B, K, L]
    float* __restrict__ out)                  // [B, K, D]
{
    __shared__ __align__(16) unsigned short hat_s[Kn * Ln * Dn]; // 102400 B, [k][l][d]
    __shared__ float cw_s[Kn * Ln];                 // 3200 B
    __shared__ float w_s[Kn * Ln];                  // 3200 B
    __shared__ float cap_s[Kn * Dn];                // 1024 B
    __shared__ float scale_s[Kn];

    const int b   = blockIdx.x;
    const int tid = threadIdx.x;

    // Async DMA: hat row slices [l][e] -> LDS [k][l][d], 16B granules.
    // Aligned groups of 4 dwords are contiguous in both layouts, so B128 works.
    {
        unsigned int* hs32 = (unsigned int*)hat_s;
        const unsigned int* hg32 = (const unsigned int*)hat;
        for (int idx4 = tid; idx4 < Ln * (En / 8); idx4 += 256) {
            const int l = idx4 >> 5;            // En/8 = 32 b128s per row
            const int j = (idx4 & 31) << 2;     // dword index in row, e = 2j
            const int k  = j >> 5;              // (2j)/64
            const int dd = j & 31;              // ((2j)%64)/2
            const unsigned long long ga =
                (unsigned long long)(hg32 + ((size_t)l * Bn + b) * (En / 2) + j);
            // LDS byte offset within the workgroup allocation (flat addr low bits)
            const unsigned int ldsoff =
                (unsigned int)(unsigned long long)(hs32 + k * (Ln * Dn / 2)
                                                        + l * (Dn / 2) + dd);
            asm volatile("global_load_async_to_lds_b128 %0, %1, off"
                         :: "v"(ldsoff), "v"(ga) : "memory");
        }
    }

    for (int idx = tid; idx < Kn * Ln; idx += 256)
        cw_s[idx] = cwg[(size_t)b * (Kn * Ln) + idx];

    asm volatile("s_wait_asynccnt 0" ::: "memory");
    __syncthreads();

    const int kk = tid >> 6;    // 0..3
    const int d  = tid & 63;    // 0..63

    float capv = 0.f;
    for (int it = 0; it < 3; ++it) {
        // softmax over K for each l
        if (tid < Ln) {
            const int l = tid;
            const float x0 = cw_s[0 * Ln + l], x1 = cw_s[1 * Ln + l];
            const float x2 = cw_s[2 * Ln + l], x3 = cw_s[3 * Ln + l];
            const float mx = fmaxf(fmaxf(x0, x1), fmaxf(x2, x3));
            const float e0v = __expf(x0 - mx), e1v = __expf(x1 - mx);
            const float e2v = __expf(x2 - mx), e3v = __expf(x3 - mx);
            const float inv = 1.f / (e0v + e1v + e2v + e3v);
            w_s[0 * Ln + l] = e0v * inv; w_s[1 * Ln + l] = e1v * inv;
            w_s[2 * Ln + l] = e2v * inv; w_s[3 * Ln + l] = e3v * inv;
        }
        __syncthreads();

        // cap[k][d] = sum_l w[k][l] * hat[k][l][d]
        float acc = 0.f;
        const unsigned short* hrow = hat_s + kk * (Ln * Dn) + d;
        const float* wrow = w_s + kk * Ln;
        #pragma unroll 4
        for (int l = 0; l < Ln; ++l)
            acc = fmaf(wrow[l], bf16u_to_f(hrow[l * Dn]), acc);

        // squash
        cap_s[tid] = acc;
        __syncthreads();
        if (tid < Kn) {
            float nsum = 0.f;
            #pragma unroll
            for (int j = 0; j < Dn; ++j) {
                const float v = cap_s[tid * Dn + j];
                nsum = fmaf(v, v, nsum);
            }
            scale_s[tid] = nsum / (1.f + nsum) / sqrtf(nsum + 1e-9f);
        }
        __syncthreads();
        capv = acc * scale_s[kk];

        if (it < 2) {
            cap_s[tid] = capv;
            __syncthreads();
            // delta[k][l] = hat[k][l][:] . cap[k][:];  cw += delta
            for (int idx = tid; idx < Kn * Ln; idx += 256) {
                const int k2 = idx / Ln, l = idx % Ln;
                const unsigned short* h2 = hat_s + k2 * (Ln * Dn) + l * Dn;
                const float* c2 = cap_s + k2 * Dn;
                float s = 0.f;
                #pragma unroll 8
                for (int j = 0; j < Dn; ++j)
                    s = fmaf(bf16u_to_f(h2[j]), c2[j], s);
                cw_s[idx] += s;
            }
            __syncthreads();
        }
    }

    out[(size_t)b * (Kn * Dn) + tid] = capv;   // [B][K][D], coalesced
}

extern "C" void kernel_launch(void* const* d_in, const int* in_sizes, int n_in,
                              void* d_out, int out_size, void* d_ws, size_t ws_size,
                              hipStream_t stream) {
    const float* seq = (const float*)d_in[0];   // [B, L, D]
    // d_in[1]: mask (unused by forward)
    const float* W   = (const float*)d_in[2];   // [1, L, K*D, D]
    const float* cw  = (const float*)d_in[3];   // [B, K, L]
    float* out = (float*)d_out;                 // [B, K, D]
    unsigned short* hat = (unsigned short*)d_ws;  // [L, B, En] bf16 (100 MB)

    dim3 g1(Bn / 16, Ln);
    hat_gemm_kernel<<<g1, 256, 0, stream>>>(seq, W, hat);
    routing_kernel<<<Bn, 256, 0, stream>>>(hat, cw, out);
}
